// DGCN_25443386262434
// MI455X (gfx1250) — matrix operand order repaired and checked by reading, not compiled, not run on hardware
//
#include <hip/hip_runtime.h>
#include <hip/hip_bf16.h>
#include <math.h>

// ---------------------------------------------------------------------------
// DGCN block for MI455X (gfx1250, wave32, WMMA).
// Key algebra: head-conv+residual+head-sum folds into q-scaling (c_h/sqrt(G));
// additive bias cancels in LayerNorm; only diag(L) is consumed downstream, so
// the N x N attention matrices stay in LDS (never hit HBM).
// Big GEMMs (x x^T and q'k^T, ~11 GFLOP) run as v_wmma_f32_16x16x32_bf16.
// ---------------------------------------------------------------------------

typedef __bf16 bf16_t;
typedef __attribute__((ext_vector_type(16))) __bf16 v16bf;
typedef __attribute__((ext_vector_type(8)))  __bf16 v8bf;
typedef __attribute__((ext_vector_type(8)))  float  v8f;

#define B_   4
#define N_   2048
#define E_   64
#define G_   64
#define H_   4
#define D_   256
#define ROWS (B_ * N_)
#define EPSL 1e-5f

__device__ __forceinline__ unsigned short f2bf(float f) {
  union { float f; unsigned u; } v; v.f = f;
  unsigned r = v.u + 0x7FFFu + ((v.u >> 16) & 1u);   // round-to-nearest-even
  return (unsigned short)(r >> 16);
}
__device__ __forceinline__ float sigmoidf_(float x) { return 1.f / (1.f + __expf(-x)); }
__device__ __forceinline__ float lrelu_(float x)    { return x > 0.f ? x : 0.01f * x; }

// ---------------------------------------------------------------------------
// Kernel 1: per-row prep. hyper fc stack -> GRU gates -> LN (Hg == `last`
// output), then q/k projections with the head-conv scale c_h/sqrt(G) folded
// into q'. Emits bf16 x', q', k panels for the WMMA kernel.
// One row per 64-thread block (2 waves).
// ---------------------------------------------------------------------------
__global__ __launch_bounds__(64)
void k1_row_prep(const float* __restrict__ xin, const float* __restrict__ lastg,
                 const float* __restrict__ fc1w, const float* __restrict__ fc1b,
                 const float* __restrict__ fc2w, const float* __restrict__ fc2b,
                 const float* __restrict__ fc3w, const float* __restrict__ fc3b,
                 const float* __restrict__ wz,  const float* __restrict__ wr,
                 const float* __restrict__ wh,
                 const float* __restrict__ bng, const float* __restrict__ bnb,
                 const float* __restrict__ qw,  const float* __restrict__ kw,
                 const float* __restrict__ mlpw,
                 unsigned short* __restrict__ xb, unsigned short* __restrict__ qb,
                 unsigned short* __restrict__ kb, float* __restrict__ out_last) {
  __shared__ float sXin[64], sLast[64], sH1[16], sH2[2], sXp[64], sRL[64], sHg[64];
  __shared__ float sRed[64], sRed2[64];
  const int t = threadIdx.x;
  const int row = blockIdx.x;                       // b*N + n
  sXin[t]  = xin[(size_t)row * G_ + t];
  sLast[t] = lastg[(size_t)row * G_ + t];
  __syncthreads();
  if (t < 16) {
    float a = fc1b[t];
    #pragma unroll 8
    for (int i = 0; i < 64; ++i) a += sXin[i] * fc1w[i * 16 + t];
    sH1[t] = sigmoidf_(a);
  }
  __syncthreads();
  if (t < 2) {
    float a = fc2b[t];
    #pragma unroll
    for (int i = 0; i < 16; ++i) a += sH1[i] * fc2w[i * 2 + t];
    sH2[t] = sigmoidf_(a);
  }
  __syncthreads();
  float xp = sH2[0] * fc3w[t] + sH2[1] * fc3w[64 + t] + fc3b[t];
  sXp[t] = xp;
  xb[(size_t)row * G_ + t] = f2bf(xp);
  __syncthreads();
  // GRU gates: tmp = [x', last]
  float az = 0.f, ar = 0.f;
  for (int i = 0; i < 64; ++i) { float v = sXp[i];   az += v * wz[i * 64 + t];        ar += v * wr[i * 64 + t]; }
  for (int i = 0; i < 64; ++i) { float v = sLast[i]; az += v * wz[(64 + i) * 64 + t]; ar += v * wr[(64 + i) * 64 + t]; }
  float z = sigmoidf_(az), r = sigmoidf_(ar);
  sRL[t] = r * sLast[t];
  __syncthreads();
  // h_t = tanh([r*last, x'] @ w_h)
  float ah = 0.f;
  for (int i = 0; i < 64; ++i) ah += sRL[i] * wh[i * 64 + t];
  for (int i = 0; i < 64; ++i) ah += sXp[i] * wh[(64 + i) * 64 + t];
  float ht  = tanhf(ah);
  float pre = (1.f - z) * sLast[t] + z * ht;
  // LN over G=64
  sRed[t] = pre; sRed2[t] = pre * pre; __syncthreads();
  for (int s = 32; s > 0; s >>= 1) {
    if (t < s) { sRed[t] += sRed[t + s]; sRed2[t] += sRed2[t + s]; }
    __syncthreads();
  }
  float m   = sRed[0] * (1.f / 64.f);
  float var = sRed2[0] * (1.f / 64.f) - m * m;
  float inv = rsqrtf(var + EPSL);
  float hg  = (pre - m) * inv * bng[t] + bnb[t];
  sHg[t] = hg;
  out_last[(size_t)row * G_ + t] = hg;              // second output (`last`)
  __syncthreads();
  // q' (folded head scale, / sqrt(G)) and k projections, leaky-relu
  #pragma unroll
  for (int j = 0; j < 4; ++j) {
    int d = t + j * 64;
    float aq = 0.f, ak = 0.f;
    for (int i = 0; i < 64; ++i) { float v = sHg[i]; aq += v * qw[i * 256 + d]; ak += v * kw[i * 256 + d]; }
    int h = d >> 6;
    float c = 1.f + mlpw[0 * 4 + h] + mlpw[1 * 4 + h] + mlpw[2 * 4 + h] + mlpw[3 * 4 + h];
    qb[(size_t)row * D_ + d] = f2bf(lrelu_(aq) * c * 0.125f);
    kb[(size_t)row * D_ + d] = f2bf(lrelu_(ak));
  }
}

// ---------------------------------------------------------------------------
// Kernel 2: fused attention. One 16-row tile per 256-thread block (8 waves).
// Waves stripe over the 128 column tiles: 2 WMMA for soc (K=64) + 8 WMMA for
// att (K=256) per 16x16 tile, results parked in LDS (two 16x2048 f32 panels,
// ~300 KB dynamic LDS, within the 320 KB/WGP budget). Then LN statistics,
// relu rowsum + diagonal -> dL = relu(adj[n,n]) / sum_m relu(adj[n,m]).
// ---------------------------------------------------------------------------
__global__ __launch_bounds__(256)
void k2_attn_fused(const unsigned short* __restrict__ xb,
                   const unsigned short* __restrict__ qb,
                   const unsigned short* __restrict__ kb,
                   const float* __restrict__ skg, const float* __restrict__ skb,
                   const float* __restrict__ atg, const float* __restrict__ atb,
                   float* __restrict__ dLout) {
  extern __shared__ char smem[];
  float* sAtt  = (float*)smem;            // 16 x 2048
  float* sSoc  = sAtt  + 16 * 2048;       // 16 x 2048
  float* sNorm = sSoc  + 16 * 2048;       // 4 x 2048  (atg, atb, skg, skb)
  float* sRed  = sNorm + 4 * 2048;        // 4 x 16 x 16 partials
  float* sStat = sRed  + 4 * 16 * 16;     // 16 x 4  (meanA, invA, meanS, invS)
  float* sDiag = sStat + 16 * 4;          // 16

  const int tid  = threadIdx.x;
  const int wave = tid >> 5, lane = tid & 31;
  const int ar = lane & 15, ah = lane >> 4;
  const int b  = blockIdx.x >> 7;
  const int n0 = (blockIdx.x & 127) << 4;

  for (int i = tid; i < 2048; i += 256) {
    sNorm[i]            = atg[i];
    sNorm[2048 + i]     = atb[i];
    sNorm[2 * 2048 + i] = skg[i];
    sNorm[3 * 2048 + i] = skb[i];
  }

  // A fragments (16 rows of this tile), contiguous K layout: lane<16 holds
  // K [0..15]; lane>=16 holds [16..31].
  const size_t baseQ = ((size_t)b * N_ + n0) * D_;
  const size_t baseX = ((size_t)b * N_ + n0) * G_;
  v16bf aQ[8], aX[2];
  #pragma unroll
  for (int kc = 0; kc < 8; ++kc) {
    const unsigned short* p = qb + baseQ + (size_t)ar * D_ + kc * 32 + ah * 16;
    v8bf lo = *(const v8bf*)(p);
    v8bf hi = *(const v8bf*)(p + 8);
    aQ[kc] = __builtin_shufflevector(lo, hi, 0,1,2,3,4,5,6,7,8,9,10,11,12,13,14,15);
  }
  #pragma unroll
  for (int kc = 0; kc < 2; ++kc) {
    const unsigned short* p = xb + baseX + (size_t)ar * G_ + kc * 32 + ah * 16;
    v8bf lo = *(const v8bf*)(p);
    v8bf hi = *(const v8bf*)(p + 8);
    aX[kc] = __builtin_shufflevector(lo, hi, 0,1,2,3,4,5,6,7,8,9,10,11,12,13,14,15);
  }

  // Column tile loop: B layout lane = column (ar), contiguous 16 K per half.
  for (int mt = wave; mt < 128; mt += 8) {
    const int m0 = mt * 16;
    const size_t rowm = (size_t)b * N_ + m0;
    if (mt + 8 < 128)
      __builtin_prefetch(kb + (rowm + 128 + ar) * D_, 0, 3);
    v8f cS = {0.f, 0.f, 0.f, 0.f, 0.f, 0.f, 0.f, 0.f};
    v8f cA = {0.f, 0.f, 0.f, 0.f, 0.f, 0.f, 0.f, 0.f};
    #pragma unroll
    for (int kc = 0; kc < 2; ++kc) {
      v16bf bx = *(const v16bf*)(xb + (rowm + ar) * G_ + kc * 32 + ah * 16);
      cS = __builtin_amdgcn_wmma_f32_16x16x32_bf16(false, aX[kc], false, bx,
                                                   (short)0, cS, false, false);
    }
    #pragma unroll
    for (int kc = 0; kc < 8; ++kc) {
      v16bf bk = *(const v16bf*)(kb + (rowm + ar) * D_ + kc * 32 + ah * 16);
      cA = __builtin_amdgcn_wmma_f32_16x16x32_bf16(false, aQ[kc], false, bk,
                                                   (short)0, cA, false, false);
    }
    // C layout: lane<16 VGPR v -> (M=v, N=lane); lane>=16 -> (M=v+8, N=lane-16)
    #pragma unroll
    for (int v = 0; v < 8; ++v) {
      int rr = v + 8 * ah;
      sAtt[rr * 2048 + m0 + ar] = cA[v];
      sSoc[rr * 2048 + m0 + ar] = cS[v];
    }
  }
  __syncthreads();

  // Per-row mean/var over the 2048 columns for both panels.
  {
    const int row = tid & 15, part = tid >> 4;       // 16 partials x 128 cols
    float sa = 0.f, qa = 0.f, ss = 0.f, qs = 0.f;
    const float* pa = sAtt + row * 2048 + part * 128;
    const float* ps = sSoc + row * 2048 + part * 128;
    for (int i = 0; i < 128; ++i) {
      float a = pa[i], s = ps[i];
      sa += a; qa += a * a; ss += s; qs += s * s;
    }
    sRed[(0 * 16 + row) * 16 + part] = sa;
    sRed[(1 * 16 + row) * 16 + part] = qa;
    sRed[(2 * 16 + row) * 16 + part] = ss;
    sRed[(3 * 16 + row) * 16 + part] = qs;
  }
  __syncthreads();
  if (tid < 16) {
    const int row = tid;
    float sa = 0.f, qa = 0.f, ss = 0.f, qs = 0.f;
    for (int p = 0; p < 16; ++p) {
      sa += sRed[(0 * 16 + row) * 16 + p];
      qa += sRed[(1 * 16 + row) * 16 + p];
      ss += sRed[(2 * 16 + row) * 16 + p];
      qs += sRed[(3 * 16 + row) * 16 + p];
    }
    float ma = sa * (1.f / 2048.f), ms = ss * (1.f / 2048.f);
    float va = qa * (1.f / 2048.f) - ma * ma;
    float vs = qs * (1.f / 2048.f) - ms * ms;
    sStat[row * 4 + 0] = ma; sStat[row * 4 + 1] = rsqrtf(va + EPSL);
    sStat[row * 4 + 2] = ms; sStat[row * 4 + 3] = rsqrtf(vs + EPSL);
  }
  __syncthreads();

  // relu pass: adj = LN(att)+LN(soc); accumulate rowsum, capture diagonal.
  {
    const int row = tid & 15, part = tid >> 4;
    const float ma = sStat[row * 4 + 0], ia = sStat[row * 4 + 1];
    const float ms = sStat[row * 4 + 2], is = sStat[row * 4 + 3];
    const int grow = n0 + row;
    const float* pa = sAtt + row * 2048;
    const float* ps = sSoc + row * 2048;
    float rs = 0.f;
    for (int i = part * 128; i < part * 128 + 128; ++i) {
      float adj = (pa[i] - ma) * ia * sNorm[i] + sNorm[2048 + i]
                + (ps[i] - ms) * is * sNorm[2 * 2048 + i] + sNorm[3 * 2048 + i];
      float rv = adj > 0.f ? adj : 0.f;
      rs += rv;
      if (i == grow) sDiag[row] = rv;
    }
    sRed[row * 16 + part] = rs;
  }
  __syncthreads();
  if (tid < 16) {
    const int row = tid;
    float rs = 0.f;
    for (int p = 0; p < 16; ++p) rs += sRed[row * 16 + p];
    dLout[(size_t)b * N_ + n0 + row] = sDiag[row] / rs;
  }
}

// ---------------------------------------------------------------------------
// Kernel 3: xo = LN(orig_x); 3x chained 64x64 matvec scaled by dL;
// out = LN(xo + h). One row per 64-thread block.
// ---------------------------------------------------------------------------
__global__ __launch_bounds__(64)
void k3_gcn(const float* __restrict__ orig,
            const float* __restrict__ xng, const float* __restrict__ xnb,
            const float* __restrict__ w1, const float* __restrict__ b1,
            const float* __restrict__ w2, const float* __restrict__ b2,
            const float* __restrict__ w3, const float* __restrict__ b3,
            const float* __restrict__ lng, const float* __restrict__ lnb,
            const float* __restrict__ dL, float* __restrict__ out) {
  __shared__ float sXo[64], sV[64], sRed[64], sRed2[64];
  const int t = threadIdx.x;
  const int row = blockIdx.x;
  float xv = orig[(size_t)row * E_ + t];
  sRed[t] = xv; sRed2[t] = xv * xv; __syncthreads();
  for (int s = 32; s > 0; s >>= 1) {
    if (t < s) { sRed[t] += sRed[t + s]; sRed2[t] += sRed2[t + s]; }
    __syncthreads();
  }
  float m = sRed[0] * (1.f / 64.f);
  float var = sRed2[0] * (1.f / 64.f) - m * m;
  float inv = rsqrtf(var + EPSL);
  float xo = (xv - m) * inv * xng[t] + xnb[t];
  sXo[t] = xo; sV[t] = xo; __syncthreads();
  const float d = dL[row];
  #pragma unroll
  for (int L = 0; L < 3; ++L) {
    const float* W  = (L == 0) ? w1 : (L == 1) ? w2 : w3;
    const float* Bb = (L == 0) ? b1 : (L == 1) ? b2 : b3;
    float a = 0.f;
    for (int i = 0; i < 64; ++i) a += sV[i] * W[i * 64 + t];
    float hn = d * a + Bb[t];
    __syncthreads(); sV[t] = hn; __syncthreads();
  }
  float o = sXo[t] + sV[t];
  sRed[t] = o; sRed2[t] = o * o; __syncthreads();
  for (int s = 32; s > 0; s >>= 1) {
    if (t < s) { sRed[t] += sRed[t + s]; sRed2[t] += sRed2[t + s]; }
    __syncthreads();
  }
  m = sRed[0] * (1.f / 64.f);
  var = sRed2[0] * (1.f / 64.f) - m * m;
  inv = rsqrtf(var + EPSL);
  out[(size_t)row * E_ + t] = (o - m) * inv * lng[t] + lnb[t];
}

// ---------------------------------------------------------------------------
extern "C" void kernel_launch(void* const* d_in, const int* in_sizes, int n_in,
                              void* d_out, int out_size, void* d_ws, size_t ws_size,
                              hipStream_t stream) {
  (void)in_sizes; (void)n_in; (void)out_size; (void)ws_size;
  const float* orig_x = (const float*)d_in[0];
  const float* x      = (const float*)d_in[1];
  const float* lastg  = (const float*)d_in[2];
  const float* fc1w = (const float*)d_in[3];  const float* fc1b = (const float*)d_in[4];
  const float* fc2w = (const float*)d_in[5];  const float* fc2b = (const float*)d_in[6];
  const float* fc3w = (const float*)d_in[7];  const float* fc3b = (const float*)d_in[8];
  const float* wz   = (const float*)d_in[9];
  const float* wr   = (const float*)d_in[10];
  const float* wh   = (const float*)d_in[11];
  const float* bng  = (const float*)d_in[12]; const float* bnb  = (const float*)d_in[13];
  const float* qw   = (const float*)d_in[14]; const float* kw   = (const float*)d_in[15];
  const float* mlpw = (const float*)d_in[16]; /* mlp_b cancels inside LN */
  const float* skg  = (const float*)d_in[18]; const float* skb  = (const float*)d_in[19];
  const float* atg  = (const float*)d_in[20]; const float* atb  = (const float*)d_in[21];
  const float* xng  = (const float*)d_in[22]; const float* xnb  = (const float*)d_in[23];
  const float* lng  = (const float*)d_in[24]; const float* lnb  = (const float*)d_in[25];
  const float* w1   = (const float*)d_in[26]; const float* b1   = (const float*)d_in[27];
  const float* w2   = (const float*)d_in[28]; const float* b2   = (const float*)d_in[29];
  const float* w3   = (const float*)d_in[30]; const float* b3   = (const float*)d_in[31];

  // workspace layout (bf16 panels + dL), ~9.5 MB
  unsigned short* xb = (unsigned short*)d_ws;           // ROWS * 64
  unsigned short* qb = xb + (size_t)ROWS * G_;          // ROWS * 256
  unsigned short* kb = qb + (size_t)ROWS * D_;          // ROWS * 256
  float* dLbuf = (float*)(kb + (size_t)ROWS * D_);      // ROWS

  float* out      = (float*)d_out;                      // [out | last]
  float* out_last = out + (size_t)ROWS * E_;

  k1_row_prep<<<ROWS, 64, 0, stream>>>(x, lastg, fc1w, fc1b, fc2w, fc2b,
                                       fc3w, fc3b, wz, wr, wh, bng, bnb,
                                       qw, kw, mlpw, xb, qb, kb, out_last);

  const size_t smem = (size_t)(2 * 16 * 2048 + 4 * 2048 + 4 * 16 * 16 + 16 * 4 + 16)
                      * sizeof(float);                  // ~299 KB < 320 KB/WGP
  k2_attn_fused<<<B_ * (N_ / 16), 256, smem, stream>>>(xb, qb, kb, skg, skb,
                                                       atg, atb, dLbuf);

  k3_gcn<<<ROWS, 64, 0, stream>>>(orig_x, xng, xnb, w1, b1, w2, b2, w3, b3,
                                  lng, lnb, dLbuf, out);
}
